// Encoder_8031588843955
// MI455X (gfx1250) — compile-verified, matrix operand order checked by reading, not run on hardware
//
#include <hip/hip_runtime.h>
#include <hip/hip_bf16.h>
#include <math.h>

typedef __bf16 bf16t;
typedef __attribute__((ext_vector_type(16))) __bf16 v16bf;
typedef __attribute__((ext_vector_type(8)))  float  v8f;
typedef __attribute__((ext_vector_type(4)))  unsigned int u32x4;
typedef __attribute__((ext_vector_type(8)))  int i32x8;
typedef __attribute__((ext_vector_type(4)))  int i32x4;

#if defined(__has_builtin)
#if __has_builtin(__builtin_amdgcn_tensor_load_to_lds)
#define HAS_TDM 1
#endif
#endif
#ifndef HAS_TDM
#define HAS_TDM 0
#endif

namespace {
constexpr int kL = 4, kB = 8, kN = 2048, kD = 512, kH = 8, kDH = 64;
constexpr int kW = 128, kS = 16, kDFF = 2048, kK1 = 9;
constexpr int kNSEG = kN / kS;   // 128
constexpr int kWIN  = kN / kW;   // 16
constexpr int kBH   = kB * kH;   // 64
constexpr int kTOK  = kB * kN;   // 16384
constexpr float kNEG   = -3.4028234663852886e+38f;
constexpr float kLn10k = 9.210340371976184f;
constexpr int kLdsStride = 40;   // 32 + 8 pad (bank-conflict-free, 80B rows)
}

__device__ inline v8f vz8() { v8f v;
#pragma unroll
  for (int i = 0; i < 8; ++i) v[i] = 0.f;
  return v; }

// Contiguous fragment load (A-layout & row-major-B^T layout), forced b128 LDS reads.
// Lane hf=0 holds K {0..7,16..23}, hf=1 holds K {8..15,24..31}.
__device__ inline v16bf frag_row(const bf16t* p, int hf) {
  union { uint4 u[2]; v16bf v; } c;
  c.u[0] = *(const uint4*)(p + hf * 8);
  c.u[1] = *(const uint4*)(p + hf * 8 + 16);
  return c.v;
}

// ---------------------------------------------------------------------------
// TDM: load a [rows=128][cols=32 bf16] tile (row stride = tk elements) from
// global into LDS with 16B pad after each 64B row -> [128][40] layout.
// Issued by one wave; tracked with TENSORcnt.
// ---------------------------------------------------------------------------
__device__ inline void tdm_tile_128x32(const bf16t* g, void* lds, int tk) {
#if HAS_TDM
  unsigned long long ga = (unsigned long long)(uintptr_t)g;
  unsigned ldsoff = (unsigned)(uintptr_t)lds;     // low 32 bits = LDS byte addr
  u32x4 g0;
  g0[0] = 1u;                                     // count = 1
  g0[1] = ldsoff;
  g0[2] = (unsigned)(ga & 0xFFFFFFFFull);
  g0[3] = (unsigned)((ga >> 32) & 0x1FFFFFFull) | (2u << 30);   // type=2
  i32x8 g1;
  g1[0] = (int)((1u << 16) | (1u << 20) | (3u << 22) | (3u << 25)); // 2B, pad on, 16DW interval, 4DW pad
  g1[1] = (int)(((unsigned)tk & 0xFFFFu) << 16);  // tensor_dim0 lo
  g1[2] = (int)(((unsigned)tk >> 16) | (128u << 16)); // dim0 hi | tensor_dim1=128
  g1[3] = (int)(32u << 16);                       // tile_dim0 = 32
  g1[4] = 128;                                    // tile_dim1 = 128
  g1[5] = tk;                                     // tensor_dim0_stride
  g1[6] = 0; g1[7] = 0;
  i32x4 gz4; gz4[0] = 0; gz4[1] = 0; gz4[2] = 0; gz4[3] = 0;
  i32x8 gz8;
#pragma unroll
  for (int i = 0; i < 8; ++i) gz8[i] = 0;
  __builtin_amdgcn_tensor_load_to_lds(g0, g1, gz4, gz4, gz8, 0);
#else
  // Fallback: one wave copies the tile with b128 loads/stores.
  int lane = threadIdx.x & 31;
  bf16t* d = (bf16t*)lds;
#pragma unroll
  for (int i = 0; i < 16; ++i) {
    int u = i * 32 + lane, r = u >> 2, co = (u & 3) * 8;
    *(uint4*)(&d[r * kLdsStride + co]) = *(const uint4*)(g + (size_t)r * tk + co);
  }
#endif
}

__device__ inline void tdm_wait0() {
#if HAS_TDM
  __builtin_amdgcn_s_wait_tensorcnt(0);
#endif
}

// ---------------------------------------------------------------------------
// Embedding + sinusoid; also emits src_word_emb output
// ---------------------------------------------------------------------------
__global__ __launch_bounds__(256)
void embed_k(const int* __restrict__ src, const float* __restrict__ emb,
             float* __restrict__ X, float* __restrict__ OutEmb) {
  int t = blockIdx.x, n = t & (kN - 1);
  int w = src[t];
  for (int d = threadIdx.x; d < kD; d += 256) {
    float e  = emb[(size_t)w * kD + d];
    float ex = (float)(2 * (d / 2)) / (float)kD;
    float angle = (float)n * __expf(-ex * kLn10k);
    float pe = ((d & 1) == 0) ? sinf(angle) : cosf(angle);
    X[(size_t)t * kD + d]      = e + pe;
    OutEmb[(size_t)t * kD + d] = e;
  }
}

// ---------------------------------------------------------------------------
// LayerNorm over D=512 -> bf16
// ---------------------------------------------------------------------------
__global__ __launch_bounds__(256)
void ln512_k(const float* __restrict__ X, const float* __restrict__ g,
             const float* __restrict__ b, bf16t* __restrict__ Y) {
  __shared__ float red[18];
  __shared__ float ws_[16];
  int t = blockIdx.x, tid = threadIdx.x;
  const float* xp = X + (size_t)t * kD;
  float a0 = xp[tid], a1 = xp[tid + 256];
  float s = a0 + a1, q = a0 * a0 + a1 * a1;
#pragma unroll
  for (int m = 1; m < 32; m <<= 1) { s += __shfl_xor(s, m, 32); q += __shfl_xor(q, m, 32); }
  int wv = tid >> 5, lane = tid & 31;
  if (lane == 0) { ws_[wv] = s; ws_[8 + wv] = q; }
  __syncthreads();
  if (tid == 0) { float ss = 0, qq = 0;
    for (int i = 0; i < 8; ++i) { ss += ws_[i]; qq += ws_[8 + i]; }
    red[16] = ss; red[17] = qq; }
  __syncthreads();
  float mean = red[16] * (1.f / kD);
  float var  = red[17] * (1.f / kD) - mean * mean;
  float rstd = rsqrtf(var + 1e-5f);
  Y[(size_t)t * kD + tid]       = (bf16t)((a0 - mean) * rstd * g[tid]       + b[tid]);
  Y[(size_t)t * kD + tid + 256] = (bf16t)((a1 - mean) * rstd * g[tid + 256] + b[tid + 256]);
}

// ---------------------------------------------------------------------------
// Rotary -> q/kv in [BH][N][64] bf16
// ---------------------------------------------------------------------------
__global__ __launch_bounds__(256)
void rotary_k(const float* __restrict__ Yq, const float* __restrict__ Ykv,
              bf16t* __restrict__ Qo, bf16t* __restrict__ KVo) {
  int lane = threadIdx.x & 31;
  size_t rid = (size_t)blockIdx.x * 8 + (threadIdx.x >> 5);  // t*8 + h
  int t = (int)(rid >> 3), h = (int)(rid & 7);
  int b = t >> 11, n = t & (kN - 1);
  const float* qp = Yq  + (size_t)t * kD + h * kDH;
  const float* kp = Ykv + (size_t)t * kD + h * kDH;
  float inv = __expf(-(float)lane * (kLn10k / 32.f));
  float f = (float)n * inv;
  float c = cosf(f), s = sinf(f);
  float q1 = qp[lane], q2 = qp[lane + 32];
  float k1 = kp[lane], k2 = kp[lane + 32];
  size_t o = ((size_t)(b * kH + h) * kN + n) * kDH;
  Qo[o + lane]       = (bf16t)((q1 * c - q2 * s) * 0.125f);
  Qo[o + lane + 32]  = (bf16t)((q2 * c + q1 * s) * 0.125f);
  KVo[o + lane]      = (bf16t)(k1 * c - k2 * s);
  KVo[o + lane + 32] = (bf16t)(k2 * c + k1 * s);
}

// ---------------------------------------------------------------------------
// Per-position layernorm of kv over DH=64
// ---------------------------------------------------------------------------
__global__ __launch_bounds__(256)
void lnkv_k(const bf16t* __restrict__ KV, const float* __restrict__ g,
            const float* __restrict__ b, bf16t* __restrict__ O) {
  int lane = threadIdx.x & 31;
  size_t rid = (size_t)blockIdx.x * 8 + (threadIdx.x >> 5);
  const bf16t* xp = KV + rid * kDH;
  float a0 = (float)xp[lane], a1 = (float)xp[lane + 32];
  float s = a0 + a1, q = a0 * a0 + a1 * a1;
#pragma unroll
  for (int m = 1; m < 32; m <<= 1) { s += __shfl_xor(s, m, 32); q += __shfl_xor(q, m, 32); }
  float mean = s * (1.f / kDH), var = q * (1.f / kDH) - mean * mean;
  float rstd = rsqrtf(var + 1e-5f);
  O[rid * kDH + lane]      = (bf16t)((a0 - mean) * rstd * g[lane]      + b[lane]);
  O[rid * kDH + lane + 32] = (bf16t)((a1 - mean) * rstd * g[lane + 32] + b[lane + 32]);
}

// ---------------------------------------------------------------------------
// Global tokens: segment softmax (S=16), weighted sum, LN(DH) -> g  (1 wave/seg)
// ---------------------------------------------------------------------------
__global__ __launch_bounds__(256)
void pkvg_k(const bf16t* __restrict__ KV, const float* __restrict__ Wp,
            const unsigned char* __restrict__ mask,
            const float* __restrict__ g, const float* __restrict__ b,
            bf16t* __restrict__ G) {
  int lane = threadIdx.x & 31;
  size_t rid = (size_t)blockIdx.x * 8 + (threadIdx.x >> 5);  // bh*NSEG + seg
  int bh = (int)(rid >> 7), seg = (int)(rid & (kNSEG - 1));
  int bb = bh >> 3;
  const bf16t* kp = KV + ((size_t)bh * kN + seg * kS) * kDH;
  float wp1 = Wp[lane], wp2 = Wp[lane + 32];
  float p[kS];
#pragma unroll
  for (int si = 0; si < kS; ++si) {
    float d = (float)kp[si * kDH + lane] * wp1 + (float)kp[si * kDH + lane + 32] * wp2;
#pragma unroll
    for (int m = 1; m < 32; m <<= 1) d += __shfl_xor(d, m, 32);
    bool keep = mask[(size_t)bb * kN + seg * kS + si] != 0;
    p[si] = keep ? d : kNEG;
  }
  float mx = kNEG;
#pragma unroll
  for (int si = 0; si < kS; ++si) mx = fmaxf(mx, p[si]);
  float sum = 0.f;
#pragma unroll
  for (int si = 0; si < kS; ++si) { p[si] = __expf(p[si] - mx); sum += p[si]; }
  float invs = 1.f / sum, g1 = 0.f, g2 = 0.f;
#pragma unroll
  for (int si = 0; si < kS; ++si) {
    float w = p[si] * invs;
    g1 += w * (float)kp[si * kDH + lane];
    g2 += w * (float)kp[si * kDH + lane + 32];
  }
  float s = g1 + g2, q = g1 * g1 + g2 * g2;
#pragma unroll
  for (int m = 1; m < 32; m <<= 1) { s += __shfl_xor(s, m, 32); q += __shfl_xor(q, m, 32); }
  float mean = s * (1.f / kDH), var = q * (1.f / kDH) - mean * mean;
  float rstd = rsqrtf(var + 1e-5f);
  G[rid * kDH + lane]      = (bf16t)((g1 - mean) * rstd * g[lane]      + b[lane]);
  G[rid * kDH + lane + 32] = (bf16t)((g2 - mean) * rstd * g[lane + 32] + b[lane + 32]);
}

// ---------------------------------------------------------------------------
// bf16 WMMA GEMM: C[M,N] = A[M,K] * B^T  (B stored [N][K] row-major).
// 128x128 block tile, 8 waves x (32x64). Double-buffered TDM staging of both
// A and B tiles (wave 0 issues; TENSORcnt + barrier synchronize).
// EPI bits: 1=bias 2=residual(f32) 8=mask-zero.
// ---------------------------------------------------------------------------
template <int EPI>
__global__ __launch_bounds__(256)
void gemm_bf16_k(const bf16t* __restrict__ A, const bf16t* __restrict__ Bm,
                 int K, int Nall,
                 const float* __restrict__ bias, const float* __restrict__ resid,
                 const unsigned char* __restrict__ mask,
                 float* __restrict__ Cf) {
  __shared__ bf16t As[2][128 * kLdsStride];
  __shared__ bf16t Bs[2][128 * kLdsStride];
  const int m0 = blockIdx.x * 128, n0 = blockIdx.y * 128;
  const int tid = threadIdx.x, lane = tid & 31, wv = tid >> 5;
  const int hf = lane >> 4, lrow = lane & 15;
  const int wm = (wv & 3) * 32, wn = (wv >> 2) * 64;
  const int nsteps = K >> 5;
  v8f acc[2][4];
#pragma unroll
  for (int i = 0; i < 2; ++i)
#pragma unroll
    for (int j = 0; j < 4; ++j) acc[i][j] = vz8();

  if (wv == 0) {
    tdm_tile_128x32(A  + (size_t)m0 * K, &As[0][0], K);
    tdm_tile_128x32(Bm + (size_t)n0 * K, &Bs[0][0], K);
    tdm_wait0();
  }
  __syncthreads();

  int buf = 0;
  for (int s = 0; s < nsteps; ++s) {
    if (wv == 0 && s + 1 < nsteps) {
      int kb = (s + 1) << 5;
      tdm_tile_128x32(A  + (size_t)m0 * K + kb, &As[buf ^ 1][0], K);
      tdm_tile_128x32(Bm + (size_t)n0 * K + kb, &Bs[buf ^ 1][0], K);
    }
    v16bf af0 = frag_row(&As[buf][(wm + lrow) * kLdsStride], hf);
    v16bf af1 = frag_row(&As[buf][(wm + 16 + lrow) * kLdsStride], hf);
#pragma unroll
    for (int j = 0; j < 4; ++j) {
      v16bf bv = frag_row(&Bs[buf][(wn + j * 16 + lrow) * kLdsStride], hf);
      acc[0][j] = __builtin_amdgcn_wmma_f32_16x16x32_bf16(false, af0, false, bv, (short)0, acc[0][j], false, false);
      acc[1][j] = __builtin_amdgcn_wmma_f32_16x16x32_bf16(false, af1, false, bv, (short)0, acc[1][j], false, false);
    }
    if (wv == 0) tdm_wait0();
    __syncthreads();
    buf ^= 1;
  }
#pragma unroll
  for (int i = 0; i < 2; ++i)
#pragma unroll
    for (int j = 0; j < 4; ++j) {
      int col = n0 + wn + j * 16 + lrow;
      float bvl = (EPI & 1) ? bias[col] : 0.f;
#pragma unroll
      for (int r = 0; r < 8; ++r) {
        int row = m0 + wm + i * 16 + r + hf * 8;
        float v = acc[i][j][r] + bvl;
        if (EPI & 2) v += resid[(size_t)row * Nall + col];
        if (EPI & 8) { if (mask[row]) v = 0.f; }
        Cf[(size_t)row * Nall + col] = v;
      }
    }
}

// ---------------------------------------------------------------------------
// conv1 (K=9 taps, D->DFF) im2col GEMM.  A: shifted vector loads (zero pad at
// sequence edges); B: TDM from tap-major weights [tap][f][d].  Bias + GELU.
// ---------------------------------------------------------------------------
__global__ __launch_bounds__(256)
void conv1_gemm_k(const bf16t* __restrict__ A, const bf16t* __restrict__ Bm,
                  const float* __restrict__ bias, bf16t* __restrict__ Cb) {
  __shared__ bf16t As[2][128 * kLdsStride];
  __shared__ bf16t Bs[2][128 * kLdsStride];
  const int m0 = blockIdx.x * 128, n0 = blockIdx.y * 128;
  const int tid = threadIdx.x, lane = tid & 31, wv = tid >> 5;
  const int hf = lane >> 4, lrow = lane & 15;
  const int wm = (wv & 3) * 32, wn = (wv >> 2) * 64;
  const int nsteps = kK1 * (kD >> 5);   // 144
  v8f acc[2][4];
#pragma unroll
  for (int i = 0; i < 2; ++i)
#pragma unroll
    for (int j = 0; j < 4; ++j) acc[i][j] = vz8();

  auto loadA = [&](int s, int bsel) {
    int tap = s >> 4, kb = (s & 15) << 5;
#pragma unroll
    for (int q = 0; q < 2; ++q) {
      int u = q * 256 + tid, r = u >> 2, co = (u & 3) * 8;
      int t = m0 + r, n = t & (kN - 1), n2 = n + tap - 4;
      uint4 val = {0u, 0u, 0u, 0u};
      if ((unsigned)n2 < (unsigned)kN)
        val = *(const uint4*)(A + (size_t)(t + tap - 4) * kD + kb + co);
      *(uint4*)(&As[bsel][r * kLdsStride + co]) = val;
    }
  };
  auto issueB = [&](int s, int bsel) {
    int tap = s >> 4, kb = (s & 15) << 5;
    tdm_tile_128x32(Bm + ((size_t)tap * kDFF + n0) * kD + kb, &Bs[bsel][0], kD);
  };

  loadA(0, 0);
  if (wv == 0) { issueB(0, 0); tdm_wait0(); }
  __syncthreads();

  int buf = 0;
  for (int s = 0; s < nsteps; ++s) {
    if (s + 1 < nsteps) {
      if (wv == 0) issueB(s + 1, buf ^ 1);
      loadA(s + 1, buf ^ 1);
    }
    v16bf af0 = frag_row(&As[buf][(wm + lrow) * kLdsStride], hf);
    v16bf af1 = frag_row(&As[buf][(wm + 16 + lrow) * kLdsStride], hf);
#pragma unroll
    for (int j = 0; j < 4; ++j) {
      v16bf bv = frag_row(&Bs[buf][(wn + j * 16 + lrow) * kLdsStride], hf);
      acc[0][j] = __builtin_amdgcn_wmma_f32_16x16x32_bf16(false, af0, false, bv, (short)0, acc[0][j], false, false);
      acc[1][j] = __builtin_amdgcn_wmma_f32_16x16x32_bf16(false, af1, false, bv, (short)0, acc[1][j], false, false);
    }
    if (wv == 0) tdm_wait0();
    __syncthreads();
    buf ^= 1;
  }
#pragma unroll
  for (int i = 0; i < 2; ++i)
#pragma unroll
    for (int j = 0; j < 4; ++j) {
      int col = n0 + wn + j * 16 + lrow;
      float bvl = bias[col];
#pragma unroll
      for (int r = 0; r < 8; ++r) {
        int row = m0 + wm + i * 16 + r + hf * 8;
        float v = acc[i][j][r] + bvl;
        v = 0.5f * v * (1.f + erff(v * 0.70710678118654752f));
        Cb[(size_t)row * kDFF + col] = (bf16t)v;
      }
    }
}

// ---------------------------------------------------------------------------
// Long-short attention core: block per (window, bh).  K=V = [128 g ; 256 local],
// WMMA QK^T -> masks -> softmax -> WMMA P V (via transposed VsT for contiguous
// B fragments).  Output merged per-head into [TOK,512] bf16.
// ---------------------------------------------------------------------------
__global__ __launch_bounds__(256)
void attn_k(const bf16t* __restrict__ Q, const bf16t* __restrict__ LK,
            const bf16t* __restrict__ G, const unsigned char* __restrict__ mask,
            bf16t* __restrict__ AO) {
  extern __shared__ char smem[];
  bf16t* Qs  = (bf16t*)smem;             // [128][72]
  bf16t* Ks  = Qs + 128 * 72;            // [384][72]   keys (rows) for QK^T
  bf16t* VsT = Ks + 384 * 72;            // [64][392]   V transposed for P@V
  bf16t* Ps  = VsT + 64 * 392;           // [128][392]
  const int win = blockIdx.x, bh = blockIdx.y;
  const int bb = bh >> 3, hh = bh & 7;
  const int tid = threadIdx.x, lane = tid & 31, wv = tid >> 5;
  const int hf = lane >> 4, lrow = lane & 15;

  const bf16t* Qg = Q + ((size_t)bh * kN + win * kW) * kDH;
#pragma unroll
  for (int q = 0; q < 4; ++q) {
    int u = q * 256 + tid, r = u >> 3, co = (u & 7) * 8;
    *(uint4*)(&Qs[r * 72 + co]) = *(const uint4*)(Qg + r * kDH + co);
  }
#pragma unroll
  for (int q = 0; q < 12; ++q) {
    int u = q * 256 + tid, r = u >> 3, co = (u & 7) * 8;
    uint4 val = {0u, 0u, 0u, 0u};
    if (r < 128) {
      val = *(const uint4*)(G + ((size_t)bh * kNSEG + r) * kDH + co);
    } else {
      int kp = (win - 1) * kW + (r - 128);
      if (kp >= 0) val = *(const uint4*)(LK + ((size_t)bh * kN + kp) * kDH + co);
    }
    *(uint4*)(&Ks[r * 72 + co]) = val;
    bf16t tmp[8];
    *(uint4*)tmp = val;
#pragma unroll
    for (int e = 0; e < 8; ++e) VsT[(co + e) * 392 + r] = tmp[e];   // transpose copy
  }
  __syncthreads();

  const int qr0 = wv * 16;
  v8f sc[24];
#pragma unroll
  for (int j = 0; j < 24; ++j) sc[j] = vz8();
#pragma unroll
  for (int kk = 0; kk < 64; kk += 32) {
    v16bf af = frag_row(&Qs[(qr0 + lrow) * 72 + kk], hf);
#pragma unroll
    for (int jt = 0; jt < 24; ++jt) {
      v16bf bv = frag_row(&Ks[(jt * 16 + lrow) * 72 + kk], hf);   // B = K^T
      sc[jt] = __builtin_amdgcn_wmma_f32_16x16x32_bf16(false, af, false, bv, (short)0, sc[jt], false, false);
    }
  }

  const unsigned char* mb = mask + (size_t)bb * kN;
#pragma unroll
  for (int r = 0; r < 8; ++r) {
    int il = qr0 + r + hf * 8;
    int qpos = win * kW + il;
    float mx = kNEG;
#pragma unroll
    for (int jt = 0; jt < 24; ++jt) {
      int j = jt * 16 + lrow;
      float s = sc[jt][r];
      bool keep;
      if (j < kNSEG) {
        keep = (qpos >= (j + 1) * kS - 1);            // causal on global tokens
      } else {
        int kp = (win - 1) * kW + (j - kNSEG);
        keep = (kp >= 0) && (kp <= qpos) && (mb[kp] != 0);
      }
      s = keep ? s : kNEG;
      sc[jt][r] = s;
      mx = fmaxf(mx, s);
    }
#pragma unroll
    for (int m = 1; m < 16; m <<= 1) mx = fmaxf(mx, __shfl_xor(mx, m, 32));
    float sum = 0.f;
#pragma unroll
    for (int jt = 0; jt < 24; ++jt) { float pp = __expf(sc[jt][r] - mx); sc[jt][r] = pp; sum += pp; }
#pragma unroll
    for (int m = 1; m < 16; m <<= 1) sum += __shfl_xor(sum, m, 32);
    float inv = 1.f / sum;
#pragma unroll
    for (int jt = 0; jt < 24; ++jt)
      Ps[il * 392 + jt * 16 + lrow] = (bf16t)(sc[jt][r] * inv);
  }
  __syncthreads();

  v8f oc[4];
#pragma unroll
  for (int j = 0; j < 4; ++j) oc[j] = vz8();
#pragma unroll
  for (int ks = 0; ks < 12; ++ks) {
    v16bf af = frag_row(&Ps[(qr0 + lrow) * 392 + ks * 32], hf);
#pragma unroll
    for (int nt = 0; nt < 4; ++nt) {
      v16bf bv = frag_row(&VsT[(nt * 16 + lrow) * 392 + ks * 32], hf);  // B = V via V^T rows
      oc[nt] = __builtin_amdgcn_wmma_f32_16x16x32_bf16(false, af, false, bv, (short)0, oc[nt], false, false);
    }
  }
#pragma unroll
  for (int nt = 0; nt < 4; ++nt)
#pragma unroll
    for (int r = 0; r < 8; ++r) {
      int il = qr0 + r + hf * 8;
      int t = bb * kN + win * kW + il;
      AO[(size_t)t * kD + hh * kDH + nt * 16 + lrow] = (bf16t)oc[nt][r];
    }
}

// ---------------------------------------------------------------------------
// Weight converters
// ---------------------------------------------------------------------------
__global__ void tobf16_k(const float* __restrict__ src, bf16t* __restrict__ dst, int n) {
  int i = blockIdx.x * 256 + threadIdx.x;
  if (i < n) dst[i] = (bf16t)src[i];
}

// w1 [f][d][tap] -> w1t [tap][f][d]   (tap-major, d contiguous: TDM-friendly B)
__global__ void w1t_k(const float* __restrict__ w1, bf16t* __restrict__ dst) {
  size_t i = (size_t)blockIdx.x * 256 + threadIdx.x;
  if (i >= (size_t)kK1 * kD * kDFF) return;
  int d = (int)(i % kD);
  size_t row = i / kD;
  int f = (int)(row % kDFF), tap = (int)(row / kDFF);
  dst[i] = (bf16t)w1[(size_t)f * (kD * kK1) + d * kK1 + tap];
}

// ---------------------------------------------------------------------------
extern "C" void kernel_launch(void* const* d_in, const int* in_sizes, int n_in,
                              void* d_out, int out_size, void* d_ws, size_t ws_size,
                              hipStream_t stream) {
  (void)in_sizes; (void)n_in; (void)out_size; (void)ws_size;
  const int*           src  = (const int*)d_in[0];
  const unsigned char* mask = (const unsigned char*)d_in[1];
  const float* emb  = (const float*)d_in[2];
  const float* ln1g = (const float*)d_in[3];
  const float* ln1b = (const float*)d_in[4];
  const float* Wq   = (const float*)d_in[5];
  const float* Wkv  = (const float*)d_in[6];
  const float* Wp   = (const float*)d_in[7];
  const float* lng  = (const float*)d_in[8];
  const float* lnb_ = (const float*)d_in[9];
  const float* gng  = (const float*)d_in[10];
  const float* gnb  = (const float*)d_in[11];
  const float* Wo   = (const float*)d_in[12];
  const float* bo   = (const float*)d_in[13];
  const float* ln2g = (const float*)d_in[14];
  const float* ln2b = (const float*)d_in[15];
  const float* w1   = (const float*)d_in[16];
  const float* b1   = (const float*)d_in[17];
  const float* w2   = (const float*)d_in[18];
  const float* b2   = (const float*)d_in[19];
  float* out_x = (float*)d_out;
  float* out_e = out_x + (size_t)kTOK * kD;

  char* p = (char*)d_ws;
  auto carve = [&](size_t bytes) { char* r = p; p += (bytes + 255) & ~(size_t)255; return r; };
  float* x   = (float*)carve((size_t)kTOK * kD * 4);
  float* yq  = (float*)carve((size_t)kTOK * kD * 4);
  float* ykv = (float*)carve((size_t)kTOK * kD * 4);   // contiguous after yq
  bf16t* lnx = (bf16t*)carve((size_t)kTOK * kD * 2);
  bf16t* qb  = (bf16t*)carve((size_t)kBH * kN * kDH * 2);
  bf16t* kvb = (bf16t*)carve((size_t)kBH * kN * kDH * 2);
  bf16t* lkb = (bf16t*)carve((size_t)kBH * kN * kDH * 2);
  bf16t* gb  = (bf16t*)carve((size_t)kBH * kNSEG * kDH * 2);
  bf16t* aob = (bf16t*)carve((size_t)kTOK * kD * 2);
  bf16t* hb  = (bf16t*)yq;   // FFN hidden aliases yq+ykv region (dead by then)
  bf16t* wqB = (bf16t*)carve((size_t)kD * kD * 2);
  bf16t* wkB = (bf16t*)carve((size_t)kD * kD * 2);
  bf16t* woB = (bf16t*)carve((size_t)kD * kD * 2);
  bf16t* w1T = (bf16t*)carve((size_t)kK1 * kD * kDFF * 2);
  bf16t* w2B = (bf16t*)carve((size_t)kDFF * kD * 2);

  embed_k<<<kTOK, 256, 0, stream>>>(src, emb, x, out_e);

  const dim3 gProj(kTOK / 128, kD / 128);     // (128, 4)
  const dim3 gFF1(kTOK / 128, kDFF / 128);    // (128, 16)
  const dim3 gAttn(kWIN, kBH);                // (16, 64)
  const size_t smemAtt = (size_t)(128 * 72 + 384 * 72 + 64 * 392 + 128 * 392) * sizeof(bf16t);

  for (int l = 0; l < kL; ++l) {
    tobf16_k<<<(kD * kD + 255) / 256, 256, 0, stream>>>(Wq  + (size_t)l * kD * kD, wqB, kD * kD);
    tobf16_k<<<(kD * kD + 255) / 256, 256, 0, stream>>>(Wkv + (size_t)l * kD * kD, wkB, kD * kD);
    tobf16_k<<<(kD * kD + 255) / 256, 256, 0, stream>>>(Wo  + (size_t)l * kD * kD, woB, kD * kD);
    w1t_k<<<(kK1 * kD * kDFF + 255) / 256, 256, 0, stream>>>(w1 + (size_t)l * kDFF * kD * kK1, w1T);
    tobf16_k<<<(kDFF * kD + 255) / 256, 256, 0, stream>>>(w2 + (size_t)l * kD * kDFF, w2B, kDFF * kD);

    ln512_k<<<kTOK, 256, 0, stream>>>(x, ln1g + l * kD, ln1b + l * kD, lnx);
    gemm_bf16_k<0><<<gProj, 256, 0, stream>>>(lnx, wqB, kD, kD, nullptr, nullptr, nullptr, yq);
    gemm_bf16_k<0><<<gProj, 256, 0, stream>>>(lnx, wkB, kD, kD, nullptr, nullptr, nullptr, ykv);
    rotary_k<<<kTOK * kH / 8, 256, 0, stream>>>(yq, ykv, qb, kvb);
    lnkv_k<<<kBH * kN / 8, 256, 0, stream>>>(kvb, lng + l * kDH, lnb_ + l * kDH, lkb);
    pkvg_k<<<kBH * kNSEG / 8, 256, 0, stream>>>(kvb, Wp + l * kDH, mask, gng + l * kDH, gnb + l * kDH, gb);
    attn_k<<<gAttn, 256, smemAtt, stream>>>(qb, lkb, gb, mask, aob);
    gemm_bf16_k<1 | 2 | 8><<<gProj, 256, 0, stream>>>(aob, woB, kD, kD, bo + l * kD, x, mask, x);

    ln512_k<<<kTOK, 256, 0, stream>>>(x, ln2g + l * kD, ln2b + l * kD, lnx);
    conv1_gemm_k<<<gFF1, 256, 0, stream>>>(lnx, w1T, b1 + l * kDFF, hb);
    gemm_bf16_k<1 | 2 | 8><<<gProj, 256, 0, stream>>>(hb, w2B, kDFF, kD, b2 + l * kD, x, mask, x);
  }

  (void)hipMemcpyAsync(out_x, x, (size_t)kTOK * kD * sizeof(float), hipMemcpyDeviceToDevice, stream);
}
